// mamba_fusion5_26371099198485
// MI455X (gfx1250) — compile-verified, matrix operand order checked by reading, not run on hardware
//
#include <hip/hip_runtime.h>
#include <cstdint>
#include <cstddef>

#define TPB 256

typedef __attribute__((ext_vector_type(16))) __bf16 v16bf;
typedef __attribute__((ext_vector_type(8)))  float  v8f;

// problem dims (fixed by reference)
#define Bb   32
#define Cc   512
#define Hh   24
#define Ww   24
#define Ll   576            // H*W
#define Dd   1024
#define Nn   4
#define Rr   32
#define Kk   4
#define MHh  2048
#define Mtok (Bb*Ll)        // 18432

// ---------- scalar helpers ----------
static __device__ __forceinline__ unsigned short f2bf(float f) {
  union { float f; unsigned u; } v; v.f = f;
  unsigned r = v.u + 0x7FFFu + ((v.u >> 16) & 1u);   // RNE
  return (unsigned short)(r >> 16);
}
static __device__ __forceinline__ float bf2f(unsigned short h) {
  union { unsigned u; float f; } v; v.u = ((unsigned)h) << 16;
  return v.f;
}
static __device__ __forceinline__ float siluf(float x) { return x / (1.0f + __expf(-x)); }
static __device__ __forceinline__ float softplusf(float x) {
  return (x > 20.0f) ? x : log1pf(__expf(x));
}
static __device__ __forceinline__ float geluf(float x) {
  float x3 = x * x * x;
  return 0.5f * x * (1.0f + tanhf(0.7978845608028654f * (x + 0.044715f * x3)));
}
// direction l -> spatial position p (row-major) ; H==W==24
static __device__ __forceinline__ int dirmap(int k, int l) {
  int ll = (k >= 2) ? (Ll - 1 - l) : l;
  return (k & 1) ? ((ll % Ww) * Ww + (ll / Ww)) : ll;
}

// ---------- gfx1250 async global -> LDS (ASYNCcnt-tracked) ----------
static __device__ __forceinline__ void async_load_b128(const void* gsrc, void* ldst) {
  unsigned l32 = (unsigned)(unsigned long long)ldst;   // low 32 bits = LDS byte offset
  asm volatile("global_load_async_to_lds_b128 %0, %1, off"
               :: "v"(l32), "v"(gsrc) : "memory");
}
template <int N>
static __device__ __forceinline__ void wait_asynccnt() {
  asm volatile("s_wait_asynccnt %0" :: "i"(N) : "memory");
}

// ---------- fp32 -> bf16 weight convert ----------
__global__ __launch_bounds__(TPB)
void k_cvt_bf16(const float* __restrict__ in, unsigned short* __restrict__ out, int n) {
  int i = blockIdx.x * TPB + threadIdx.x;
  if (i < n) out[i] = f2bf(in[i]);
}

// ---------- fuse 3 modalities with patch->token transpose (LDS tiled) ----------
__global__ __launch_bounds__(TPB)
void k_fuse(const float* __restrict__ x1, const float* __restrict__ x2,
            const float* __restrict__ x3, float* __restrict__ xt) {
  __shared__ float tile[64][33];
  int ct = blockIdx.x, lt = blockIdx.y, b = blockIdx.z;
  int c0 = ct * 64, l0 = lt * 32;
  int tid = threadIdx.x;
  #pragma unroll
  for (int q = 0; q < 8; q++) {
    int idx = tid + q * TPB;
    int cc = idx >> 5, ll = idx & 31;              // coalesced along l
    size_t g = ((size_t)(b * Cc + c0 + cc)) * Ll + (l0 + ll);
    tile[cc][ll] = x1[g] + x2[g] + x3[g];
  }
  __syncthreads();
  #pragma unroll
  for (int q = 0; q < 8; q++) {
    int idx = tid + q * TPB;
    int cw = idx & 63, lw = idx >> 6;              // coalesced along c
    xt[((size_t)(b * Ll + l0 + lw)) * Cc + (c0 + cw)] = tile[cw][lw];
  }
}

// ---------- token->patch transpose:  out(B,C,L) = in(B,L,C) ----------
__global__ __launch_bounds__(TPB)
void k_t2p(const float* __restrict__ in, float* __restrict__ out) {
  __shared__ float tile[64][33];
  int ct = blockIdx.x, lt = blockIdx.y, b = blockIdx.z;
  int c0 = ct * 64, l0 = lt * 32;
  int tid = threadIdx.x;
  #pragma unroll
  for (int q = 0; q < 8; q++) {
    int idx = tid + q * TPB;
    int cw = idx & 63, lw = idx >> 6;              // coalesced along c
    tile[cw][lw] = in[((size_t)(b * Ll + l0 + lw)) * Cc + (c0 + cw)];
  }
  __syncthreads();
  #pragma unroll
  for (int q = 0; q < 8; q++) {
    int idx = tid + q * TPB;
    int cc = idx >> 5, ll = idx & 31;              // coalesced along l
    out[((size_t)(b * Cc + c0 + cc)) * Ll + (l0 + ll)] = tile[cc][ll];
  }
}

// ---------- LayerNorm (optional silu-gate) -> bf16 ----------
__global__ __launch_bounds__(TPB)
void k_ln_gate(const float* __restrict__ in, const float* __restrict__ w,
               const float* __restrict__ bsh, const unsigned short* __restrict__ zgate,
               unsigned short* __restrict__ out, int width) {
  int t = blockIdx.x;
  int tid = threadIdx.x;
  const float* row = in + (size_t)t * width;
  float s = 0.f, s2 = 0.f;
  for (int c = tid; c < width; c += TPB) { float v = row[c]; s += v; s2 += v * v; }
  for (int off = 16; off > 0; off >>= 1) {
    s  += __shfl_down(s,  off, 32);
    s2 += __shfl_down(s2, off, 32);
  }
  __shared__ float ps[8], ps2[8], stat[2];
  int lane = tid & 31, wv = tid >> 5;
  if (lane == 0) { ps[wv] = s; ps2[wv] = s2; }
  __syncthreads();
  if (tid == 0) {
    float a = 0.f, b2 = 0.f;
    #pragma unroll
    for (int i = 0; i < 8; i++) { a += ps[i]; b2 += ps2[i]; }
    float mu = a / (float)width;
    float var = b2 / (float)width - mu * mu;
    stat[0] = mu; stat[1] = rsqrtf(var + 1e-5f);
  }
  __syncthreads();
  float mu = stat[0], rstd = stat[1];
  for (int c = tid; c < width; c += TPB) {
    float v = (row[c] - mu) * rstd * w[c] + bsh[c];
    if (zgate) v *= siluf(bf2f(zgate[(size_t)t * width + c]));
    out[(size_t)t * width + c] = f2bf(v);
  }
}

// ---------- depthwise 3x3 conv + bias + silu ; (B,L,D) layout, coalesced on d ----------
__global__ __launch_bounds__(TPB)
void k_dwconv(const float* __restrict__ xin, const float* __restrict__ cw,
              const float* __restrict__ cb, float* __restrict__ xc) {
  int p = blockIdx.x % Ll, b = blockIdx.x / Ll;
  int h = p / Ww, w = p % Ww;
  int tid = threadIdx.x;
  #pragma unroll
  for (int q = 0; q < Dd / TPB; q++) {
    int d = tid + q * TPB;
    float acc = cb[d];
    #pragma unroll
    for (int ky = 0; ky < 3; ky++) {
      int hh = h + ky - 1;
      if (hh < 0 || hh >= Hh) continue;
      #pragma unroll
      for (int kx = 0; kx < 3; kx++) {
        int ww2 = w + kx - 1;
        if (ww2 < 0 || ww2 >= Ww) continue;
        acc += cw[d * 9 + ky * 3 + kx] *
               xin[((size_t)(b * Ll + hh * Ww + ww2)) * Dd + d];
      }
    }
    xc[((size_t)(b * Ll + p)) * Dd + d] = siluf(acc);
  }
}

// ---------- y init:  y[b,p,d] = xc[b,p,d] * sum_k Ds[k,d] ----------
__global__ __launch_bounds__(TPB)
void k_yinit(const float* __restrict__ xc, const float* __restrict__ Ds,
             float* __restrict__ y) {
  size_t i = (size_t)blockIdx.x * TPB + threadIdx.x;
  int d = (int)(i % Dd);
  float sd = Ds[d] + Ds[Dd + d] + Ds[2 * Dd + d] + Ds[3 * Dd + d];
  y[i] = xc[i] * sd;
}

// ---------- x_dbl[b,k,l,0:40] = x_proj_w[k] @ xc[b, p_k(l), :] ----------
__global__ __launch_bounds__(TPB)
void k_xdbl(const float* __restrict__ xc, const float* __restrict__ xpw,
            float* __restrict__ xdbl) {
  __shared__ float v[Dd];
  int l = blockIdx.x, k = blockIdx.y, b = blockIdx.z;
  int tid = threadIdx.x;
  int p = dirmap(k, l);
  const float* src = xc + ((size_t)(b * Ll + p)) * Dd;
  #pragma unroll
  for (int q = 0; q < Dd / TPB; q++) v[tid + q * TPB] = src[tid + q * TPB];
  __syncthreads();
  int wv = tid >> 5, lane = tid & 31;
  int c0 = wv * 5;                       // 8 waves x 5 channels = 40
  float partial[5] = {0.f, 0.f, 0.f, 0.f, 0.f};
  for (int d = lane; d < Dd; d += 32) {
    float x = v[d];
    #pragma unroll
    for (int j = 0; j < 5; j++)
      partial[j] += xpw[(size_t)(k * 40 + c0 + j) * Dd + d] * x;
  }
  #pragma unroll
  for (int j = 0; j < 5; j++) {
    float pj = partial[j];
    for (int off = 16; off > 0; off >>= 1) pj += __shfl_down(pj, off, 32);
    if (lane == 0)
      xdbl[((size_t)((b * Kk + k) * Ll + l)) * 40 + c0 + j] = pj;
  }
}

// ---------- selective scan; each block owns (b, 256 d's) for ALL 4 dirs (no atomics) ----------
__global__ __launch_bounds__(TPB)
void k_scan(const float* __restrict__ xdbl, const float* __restrict__ xc,
            const float* __restrict__ A_logs, const float* __restrict__ dt_w,
            const float* __restrict__ dt_b, float* __restrict__ y) {
  __shared__ float rows[32 * 40];
  int b = blockIdx.y;
  int d = blockIdx.x * TPB + threadIdx.x;
  int tid = threadIdx.x;
  for (int k = 0; k < Kk; k++) {
    float Aloc[Nn], hst[Nn];
    #pragma unroll
    for (int n = 0; n < Nn; n++) {
      Aloc[n] = -__expf(A_logs[(size_t)(k * Dd + d) * Nn + n]);
      hst[n] = 0.f;
    }
    float dtw[Rr];
    #pragma unroll
    for (int r = 0; r < Rr; r++) dtw[r] = dt_w[(size_t)(k * Dd + d) * Rr + r];
    float dtb = dt_b[k * Dd + d];
    const float* xrow = xdbl + ((size_t)(b * Kk + k)) * Ll * 40;
    for (int l0 = 0; l0 < Ll; l0 += 32) {
      __syncthreads();
      #pragma unroll
      for (int q = 0; q < 5; q++) {                    // 32 rows * 40 = 1280 = 5*256
        int idx = tid + q * TPB;
        rows[idx] = xrow[(size_t)l0 * 40 + idx];
      }
      __syncthreads();
      for (int li = 0; li < 32; li++) {
        const float* rr = &rows[li * 40];
        float dtr = dtb;
        #pragma unroll
        for (int r = 0; r < Rr; r++) dtr += rr[r] * dtw[r];
        float dt = softplusf(dtr);
        int p = dirmap(k, l0 + li);                    // uniform -> SALU
        size_t gidx = ((size_t)(b * Ll + p)) * Dd + d;
        float u = xc[gidx];
        float du = dt * u;
        float yv = 0.f;
        #pragma unroll
        for (int n = 0; n < Nn; n++) {
          hst[n] = hst[n] * __expf(dt * Aloc[n]) + du * rr[Rr + n];
          yv += hst[n] * rr[Rr + Nn + n];
        }
        y[gidx] += yv;                                 // exclusive ownership: deterministic
      }
    }
  }
}

// ---------- WMMA bf16 GEMM, async-to-LDS double-buffered ----------
enum { EPI_INPROJ = 0, EPI_OUTPROJ = 1, EPI_FC1 = 2 };

union FragBF { v16bf v; unsigned u[8]; };

template <int EPI>
__global__ __launch_bounds__(TPB)
void k_gemm_bf16(const unsigned short* __restrict__ A, const unsigned short* __restrict__ Bw,
                 const float* __restrict__ bias, const float* __restrict__ resid,
                 float* __restrict__ outF, unsigned short* __restrict__ outH,
                 int M, int N, int Kd) {
  constexpr int BM = 128, BN = 128, BK = 32, LDT = 40;   // LDT halfs: pad vs bank conflicts
  __shared__ unsigned short sA[2][BM * LDT];
  __shared__ unsigned short sB[2][BN * LDT];
  int tid = threadIdx.x;
  int n0 = blockIdx.x * BN, m0 = blockIdx.y * BM;
  int lane = tid & 31, wv = tid >> 5;
  int wm = wv >> 2, wn = wv & 3;                         // 2x4 wave grid -> 64x32 per wave
  int r = lane & 15, hi = lane >> 4;

  v8f acc[4][2];
  #pragma unroll
  for (int a = 0; a < 4; a++)
    #pragma unroll
    for (int c2 = 0; c2 < 2; c2++)
      #pragma unroll
      for (int i = 0; i < 8; i++) acc[a][c2][i] = 0.f;

  // per-thread tile-load slots: 128 rows x 32 halfs (16B each), for both A and B
  int ldrow0 = tid >> 2, ldcol = (tid & 3) * 8;
  int ldrow1 = (tid + TPB) >> 2, ldcol1 = ((tid + TPB) & 3) * 8;

  // issue one K-tile's async loads into LDS buffer `buf` (4 async ops per wave)
  auto issue = [&](int k0, int buf) {
    async_load_b128(A + (size_t)(m0 + ldrow0) * Kd + k0 + ldcol,  &sA[buf][ldrow0 * LDT + ldcol]);
    async_load_b128(Bw + (size_t)(n0 + ldrow0) * Kd + k0 + ldcol, &sB[buf][ldrow0 * LDT + ldcol]);
    async_load_b128(A + (size_t)(m0 + ldrow1) * Kd + k0 + ldcol1, &sA[buf][ldrow1 * LDT + ldcol1]);
    async_load_b128(Bw + (size_t)(n0 + ldrow1) * Kd + k0 + ldcol1,&sB[buf][ldrow1 * LDT + ldcol1]);
  };

  int nk = Kd / BK;
  issue(0, 0);
  for (int kt = 0; kt < nk; ++kt) {
    int cur = kt & 1;
    if (kt + 1 < nk) {                      // prefetch next tile into the other buffer
      issue((kt + 1) * BK, cur ^ 1);
      wait_asynccnt<4>();                   // current tile done (async loads complete in order)
    } else {
      wait_asynccnt<0>();
    }
    __syncthreads();                        // all waves' tiles visible

    FragBF af[4], bfv[2];
    #pragma unroll
    for (int mt = 0; mt < 4; mt++) {
      const unsigned short* pr = &sA[cur][(wm * 64 + mt * 16 + r) * LDT];
      #pragma unroll
      for (int q2 = 0; q2 < 4; q2++) {
        af[mt].u[q2]     = *(const unsigned*)(pr + hi * 8 + 2 * q2);        // K 0..7 / 8..15
        af[mt].u[4 + q2] = *(const unsigned*)(pr + 16 + hi * 8 + 2 * q2);   // K 16..23 / 24..31
      }
    }
    #pragma unroll
    for (int nt = 0; nt < 2; nt++) {
      const unsigned short* pr = &sB[cur][(wn * 32 + nt * 16 + r) * LDT];
      #pragma unroll
      for (int q2 = 0; q2 < 4; q2++) {
        bfv[nt].u[q2]     = *(const unsigned*)(pr + hi * 8 + 2 * q2);
        bfv[nt].u[4 + q2] = *(const unsigned*)(pr + 16 + hi * 8 + 2 * q2);
      }
    }
    #pragma unroll
    for (int mt = 0; mt < 4; mt++)
      #pragma unroll
      for (int nt = 0; nt < 2; nt++)
        acc[mt][nt] = __builtin_amdgcn_wmma_f32_16x16x32_bf16(
            false, af[mt].v, false, bfv[nt].v, (short)0, acc[mt][nt], false, false);

    __syncthreads();                        // done reading `cur`; safe to refill next iter
  }

  // epilogue: C/D layout -> row = m0+wm*64+mt*16 + hi*8 + i ; col = n0+wn*32+nt*16 + r
  #pragma unroll
  for (int mt = 0; mt < 4; mt++)
    #pragma unroll
    for (int nt = 0; nt < 2; nt++)
      #pragma unroll
      for (int i = 0; i < 8; i++) {
        int row = m0 + wm * 64 + mt * 16 + hi * 8 + i;
        int col = n0 + wn * 32 + nt * 16 + r;
        float c = acc[mt][nt][i] + bias[col];
        if (EPI == EPI_INPROJ) {
          if (col < Dd) outF[(size_t)row * Dd + col] = c;                     // xin fp32
          else          outH[(size_t)row * Dd + (col - Dd)] = f2bf(c);        // z bf16
        } else if (EPI == EPI_OUTPROJ) {
          outF[(size_t)row * N + col] = c + resid[(size_t)row * N + col];     // + residual
        } else {                                                              // FC1
          outH[(size_t)row * N + col] = f2bf(geluf(c));                       // gelu -> bf16
        }
      }
}

// ---------- host side ----------
extern "C" void kernel_launch(void* const* d_in, const int* in_sizes, int n_in,
                              void* d_out, int out_size, void* d_ws, size_t ws_size,
                              hipStream_t stream) {
  (void)in_sizes; (void)n_in; (void)out_size; (void)ws_size;
  const float* x1        = (const float*)d_in[0];
  const float* x2        = (const float*)d_in[1];
  const float* x3        = (const float*)d_in[2];
  const float* ln1_w     = (const float*)d_in[3];
  const float* ln1_b     = (const float*)d_in[4];
  const float* in_proj_w = (const float*)d_in[5];
  const float* in_proj_b = (const float*)d_in[6];
  const float* conv_w    = (const float*)d_in[7];
  const float* conv_b    = (const float*)d_in[8];
  const float* x_proj_w  = (const float*)d_in[9];
  const float* dt_w      = (const float*)d_in[10];
  const float* dt_b      = (const float*)d_in[11];
  const float* A_logs    = (const float*)d_in[12];
  const float* Ds        = (const float*)d_in[13];
  const float* onorm_w   = (const float*)d_in[14];
  const float* onorm_b   = (const float*)d_in[15];
  const float* out_proj_w= (const float*)d_in[16];
  const float* out_proj_b= (const float*)d_in[17];
  const float* ln2_w     = (const float*)d_in[18];
  const float* ln2_b     = (const float*)d_in[19];
  const float* fc1_w     = (const float*)d_in[20];
  const float* fc1_b     = (const float*)d_in[21];
  const float* fc2_w     = (const float*)d_in[22];
  const float* fc2_b     = (const float*)d_in[23];
  float* out = (float*)d_out;

  char* ws = (char*)d_ws;
  size_t off = 0;
  auto alloc = [&](size_t bytes) -> char* {
    char* p = ws + off;
    off += (bytes + 255) & ~(size_t)255;
    return p;
  };
  float*          xt    = (float*)alloc((size_t)Mtok * Cc * 4);        // residual 1 -> fc2 tmp
  float*          xres  = (float*)alloc((size_t)Mtok * Cc * 4);        // residual 2
  unsigned short* bfA   = (unsigned short*)alloc((size_t)Mtok * Dd * 2); // h -> ygate -> m
  unsigned short* zbuf  = (unsigned short*)alloc((size_t)Mtok * Dd * 2);
  char*           big1  = alloc((size_t)Mtok * Dd * 4);                // xin -> y -> g(bf16)
  float*          xc    = (float*)alloc((size_t)Mtok * Dd * 4);
  float*          xdbl  = (float*)alloc((size_t)Bb * Kk * Ll * 40 * 4);
  unsigned short* wip   = (unsigned short*)alloc((size_t)2 * Dd * Cc * 2);
  unsigned short* wop   = (unsigned short*)alloc((size_t)Cc * Dd * 2);
  unsigned short* wf1   = (unsigned short*)alloc((size_t)MHh * Cc * 2);
  unsigned short* wf2   = (unsigned short*)alloc((size_t)Cc * MHh * 2);

  float*          xin   = (float*)big1;
  float*          ybuf  = (float*)big1;
  unsigned short* gbuf  = (unsigned short*)big1;

  // weight conversions (fp32 -> bf16)
  k_cvt_bf16<<<(2 * Dd * Cc) / TPB, TPB, 0, stream>>>(in_proj_w, wip, 2 * Dd * Cc);
  k_cvt_bf16<<<(Cc * Dd) / TPB, TPB, 0, stream>>>(out_proj_w, wop, Cc * Dd);
  k_cvt_bf16<<<(MHh * Cc) / TPB, TPB, 0, stream>>>(fc1_w, wf1, MHh * Cc);
  k_cvt_bf16<<<(Cc * MHh) / TPB, TPB, 0, stream>>>(fc2_w, wf2, Cc * MHh);

  // xt = p2t(x1)+p2t(x2)+p2t(x3)
  k_fuse<<<dim3(Cc / 64, Ll / 32, Bb), TPB, 0, stream>>>(x1, x2, x3, xt);
  // h = LN1(xt) -> bf16
  k_ln_gate<<<Mtok, TPB, 0, stream>>>(xt, ln1_w, ln1_b, nullptr, bfA, Cc);
  // in_proj GEMM: xz = h @ Wip^T + b ; split -> xin(fp32), z(bf16)
  k_gemm_bf16<EPI_INPROJ><<<dim3((2 * Dd) / 128, Mtok / 128), TPB, 0, stream>>>(
      bfA, wip, in_proj_b, nullptr, xin, zbuf, Mtok, 2 * Dd, Cc);
  // depthwise conv + silu
  k_dwconv<<<Bb * Ll, TPB, 0, stream>>>(xin, conv_w, conv_b, xc);
  // y = xc * sum_k Ds   (merge init; xin dead, reuse big1)
  k_yinit<<<((size_t)Mtok * Dd) / TPB, TPB, 0, stream>>>(xc, Ds, ybuf);
  // x_dbl per direction
  k_xdbl<<<dim3(Ll, Kk, Bb), TPB, 0, stream>>>(xc, x_proj_w, xdbl);
  // selective scan over 4 directions, accumulate into y
  k_scan<<<dim3(Dd / TPB, Bb), TPB, 0, stream>>>(xdbl, xc, A_logs, dt_w, dt_b, ybuf);
  // ygate = out_norm(y) * silu(z) -> bf16
  k_ln_gate<<<Mtok, TPB, 0, stream>>>(ybuf, onorm_w, onorm_b, zbuf, bfA, Dd);
  // out_proj GEMM + bias + xt residual -> xres fp32
  k_gemm_bf16<EPI_OUTPROJ><<<dim3(Cc / 128, Mtok / 128), TPB, 0, stream>>>(
      bfA, wop, out_proj_b, xt, xres, nullptr, Mtok, Cc, Dd);
  // m = LN2(xres) -> bf16
  k_ln_gate<<<Mtok, TPB, 0, stream>>>(xres, ln2_w, ln2_b, nullptr, bfA, Cc);
  // fc1 GEMM + gelu -> g bf16 (y dead, reuse big1)
  k_gemm_bf16<EPI_FC1><<<dim3(MHh / 128, Mtok / 128), TPB, 0, stream>>>(
      bfA, wf1, fc1_b, nullptr, nullptr, gbuf, Mtok, MHh, Cc);
  // fc2 GEMM + bias + xres residual -> row-major tmp (reuse xt; coalesced stores)
  k_gemm_bf16<EPI_OUTPROJ><<<dim3(Cc / 128, Mtok / 128), TPB, 0, stream>>>(
      gbuf, wf2, fc2_b, xres, xt, nullptr, Mtok, Cc, MHh);
  // token2patch transpose -> d_out (B,C,H,W)
  k_t2p<<<dim3(Cc / 64, Ll / 32, Bb), TPB, 0, stream>>>(xt, out);
}
